// TruePathPreservingCNN_69741678952833
// MI455X (gfx1250) — compile-verified, matrix operand order checked by reading, not run on hardware
//
#include <hip/hip_runtime.h>
#include <hip/hip_fp16.h>

typedef __attribute__((ext_vector_type(16))) _Float16 v16h;
typedef __attribute__((ext_vector_type(8)))  _Float16 v8h;
typedef __attribute__((ext_vector_type(8)))  float    v8f;

// -------------------------------------------------------------------------
// Implicit-GEMM 3x3 conv (pad=1, stride=1), fused bias + ReLU.
//   A: f16 weights [Cout, K], K = 9*CINP, k = rs*CINP + ci  (LDS-staged)
//   B: f16 zero-padded NHWC input [Nimg, H+2, W+2, CINP]
// Each 32-K WMMA step sits inside one rs => (r,s) uniform, lane's 16 B
// elements are two contiguous 16B chunks -> 2x global_load_b128.
// One wave -> 16(Cout) x 16(pixel); block = 8 waves (8 pixel tiles).
// -------------------------------------------------------------------------
template <int CINP, int H, int W>
__global__ void __launch_bounds__(256)
conv3x3_wmma_t(const _Float16* __restrict__ Wf,   // [Cout, K]
               const float*    __restrict__ bias, // [Cout]
               const _Float16* __restrict__ Xp,   // [Nimg, H+2, W+2, CINP]
               float*          __restrict__ Y,    // [Nimg, Cout, H, W]
               int Cout)
{
  constexpr int K  = 9 * CINP;
  constexpr int Wp = W + 2;
  constexpr int Hp = H + 2;
  constexpr int HW = H * W;

  extern __shared__ _Float16 sW[];                // [16][K]
  const int lane  = threadIdx.x & 31;
  const int wave  = threadIdx.x >> 5;
  const int half8 = (lane >> 4) << 3;
  const int lm    = lane & 15;

  // stage this block's 16 weight rows: 16B chunks, coalesced
  {
    const v8h* gs = (const v8h*)(Wf + (size_t)blockIdx.y * 16 * K);
    v8h* ss = (v8h*)sW;
    for (int t = threadIdx.x; t < 2 * K; t += 256) ss[t] = gs[t];
  }
  __syncthreads();

  const int pixel = (blockIdx.x * 8 + wave) * 16 + lm;
  const int img   = pixel / HW;                   // compile-time pow2 -> shifts
  const int rem   = pixel - img * HW;
  const int h     = rem / W;
  const int w     = rem - h * W;

  // output pixel (h,w) reads padded rows h..h+2, cols w..w+2
  const _Float16* xrow = Xp + ((size_t)(img * Hp + h) * Wp + w) * CINP;
  const _Float16* wrA  = sW + (size_t)lm * K + half8;

  v8f c = {};
  int k = 0;
#pragma unroll
  for (int rs = 0; rs < 9; ++rs) {
    constexpr int dummy = 0; (void)dummy;
    const int r = rs / 3, s = rs - (rs / 3) * 3;  // compile-time after unroll
    const _Float16* bp = xrow + (size_t)(r * Wp + s) * CINP + half8;
    __builtin_prefetch((const void*)(bp + CINP), 0, 1);  // global_prefetch_b8
#pragma unroll
    for (int ci0 = 0; ci0 < CINP; ci0 += 32, k += 32) {
      const v8h alo = *(const v8h*)(wrA + k);          // ds_load_b128
      const v8h ahi = *(const v8h*)(wrA + k + 16);
      const v8h blo = *(const v8h*)(bp + ci0);          // global_load_b128
      const v8h bhi = *(const v8h*)(bp + ci0 + 16);
      v16h a, b;
#pragma unroll
      for (int i = 0; i < 8; ++i) {
        a[i] = alo[i]; a[i + 8] = ahi[i];
        b[i] = blo[i]; b[i + 8] = bhi[i];
      }
      c = __builtin_amdgcn_wmma_f32_16x16x32_f16(false, a, false, b,
                                                 (short)0, c, false, false);
    }
  }

#pragma unroll
  for (int r = 0; r < 8; ++r) {
    const int co = blockIdx.y * 16 + r + half8;
    float v = c[r] + bias[co];
    v = fmaxf(v, 0.0f);
    Y[((size_t)img * Cout + co) * HW + h * W + w] = v;
  }
}

// -------------------------------------------------------------------------
// GEMM: C[M,N] = A[M,K] * B[N,K]^T (+bias, opt ReLU). f16 in, f16/f32 out.
// K % 256 == 0. Block = 8 waves tiling M, sharing one 16xK B panel staged
// through LDS in 8KB chunks. Operand fetches are 16B chunks.
// -------------------------------------------------------------------------
__global__ void __launch_bounds__(256)
gemm_nt_wmma(const _Float16* __restrict__ A,
             const _Float16* __restrict__ B,
             const float*    __restrict__ bias,
             float*          __restrict__ outF,
             _Float16*       __restrict__ outH,
             int M, int N, int K, int relu)
{
  __shared__ _Float16 sB[16 * 256];   // [n 0..15][k 0..255]
  const int lane  = threadIdx.x & 31;
  const int wave  = threadIdx.x >> 5;
  const int half8 = (lane >> 4) << 3;
  const int lm    = lane & 15;

  const int mTile = blockIdx.y * 8 + wave;
  const int nTile = blockIdx.x;

  const _Float16* arow = A + (size_t)(mTile * 16 + lm) * K + half8;
  const _Float16* bpan = B + (size_t)(nTile * 16) * K;
  const _Float16* brow = sB + (size_t)lm * 256 + half8;

  v8f c = {};
  for (int k0 = 0; k0 < K; k0 += 256) {
    // stage 16x256 B chunk: 16B per thread-iteration, coalesced
    {
      v8h* ss = (v8h*)sB;
      for (int t = threadIdx.x; t < 16 * 32; t += 256) {
        const int rr = t >> 5, cc = t & 31;
        ss[t] = *(const v8h*)(bpan + (size_t)rr * K + k0 + cc * 8);
      }
    }
    __syncthreads();
#pragma unroll
    for (int kk = 0; kk < 256; kk += 32) {
      const v8h alo = *(const v8h*)(arow + k0 + kk);
      const v8h ahi = *(const v8h*)(arow + k0 + kk + 16);
      const v8h blo = *(const v8h*)(brow + kk);
      const v8h bhi = *(const v8h*)(brow + kk + 16);
      v16h a, b;
#pragma unroll
      for (int i = 0; i < 8; ++i) {
        a[i] = alo[i]; a[i + 8] = ahi[i];
        b[i] = blo[i]; b[i + 8] = bhi[i];
      }
      c = __builtin_amdgcn_wmma_f32_16x16x32_f16(false, a, false, b,
                                                 (short)0, c, false, false);
    }
    __syncthreads();
  }

#pragma unroll
  for (int r = 0; r < 8; ++r) {
    const int gm = mTile * 16 + r + half8;
    const int gn = nTile * 16 + lm;
    float v = c[r] + bias[gn];
    if (relu) v = fmaxf(v, 0.0f);
    if (outH) outH[(size_t)gm * N + gn] = (_Float16)v;
    else      outF[(size_t)gm * N + gn] = v;
  }
}

// ---- conv weight repack: dst[co][rs*CinP+ci] = w[co][ci][rs] (f16, pad ci) ----
__global__ void cvt_conv_w(const float* __restrict__ w, _Float16* __restrict__ dst,
                           int Cout, int Cin, int CinP)
{
  const int K = 9 * CinP;
  long i = (long)blockIdx.x * blockDim.x + threadIdx.x;
  const long tot = (long)Cout * K;
  if (i >= tot) return;
  const int co = (int)(i / K);
  const int kr = (int)(i - (long)co * K);
  const int rs = kr / CinP;
  const int ci = kr - rs * CinP;
  dst[i] = (ci < Cin) ? (_Float16)w[((long)co * Cin + ci) * 9 + rs] : (_Float16)0.f;
}

// ---- fused gather(top-k paths) + zero-pad + f32->f16, NCHW -> padded NHWC ----
// out[(b*keep+j), hh, ww, ci]; in viewed [B, P, Cin, H, W]; idx==null -> j.
__global__ void gather_pad_f16(const float* __restrict__ in, const int* __restrict__ idx,
                               _Float16* __restrict__ out,
                               int B, int P, int keep, int Cin, int CinP, int H, int W)
{
  const int Wp = W + 2, Hp = H + 2;
  long i = (long)blockIdx.x * blockDim.x + threadIdx.x;
  const long tot = (long)B * keep * Hp * Wp * CinP;
  if (i >= tot) return;
  long t = i;
  const int ci = (int)(t % CinP); t /= CinP;
  const int ww = (int)(t % Wp);   t /= Wp;
  const int hh = (int)(t % Hp);   t /= Hp;
  const int j  = (int)(t % keep);
  const int b  = (int)(t / keep);
  _Float16 v = (_Float16)0.f;
  if (ci < Cin && hh >= 1 && hh <= H && ww >= 1 && ww <= W) {
    const int p = idx ? idx[b * keep + j] : j;
    v = (_Float16)in[((((long)b * P + p) * Cin + ci) * H + (hh - 1)) * W + (ww - 1)];
  }
  out[i] = v;
}

// ---- f32 -> f16 with zero padding: dst[Rp,Cp] <- src[Rr,Cr] ----
__global__ void cvt_pad_f16(const float* __restrict__ src, _Float16* __restrict__ dst,
                            int Rr, int Cr, int Rp, int Cp)
{
  long i = (long)blockIdx.x * blockDim.x + threadIdx.x;
  const long tot = (long)Rp * Cp;
  if (i >= tot) return;
  const int r = (int)(i / Cp);
  const int c = (int)(i - (long)r * Cp);
  dst[i] = (r < Rr && c < Cr) ? (_Float16)src[(long)r * Cr + c] : (_Float16)0.f;
}

__global__ void pad_bias_f32(const float* __restrict__ src, float* __restrict__ dst,
                             int nReal, int nPad)
{
  int i = blockIdx.x * blockDim.x + threadIdx.x;
  if (i < nPad) dst[i] = (i < nReal) ? src[i] : 0.f;
}

// ---- 2x2 max pool (inputs already ReLU'd by conv epilogue) ----
__global__ void pool2x2(const float* __restrict__ in, float* __restrict__ out,
                        long NC, int H, int W)
{
  const int Ho = H >> 1, Wo = W >> 1;
  long i = (long)blockIdx.x * blockDim.x + threadIdx.x;
  const long tot = NC * Ho * Wo;
  if (i >= tot) return;
  const long nc = i / (Ho * Wo);
  const int  r  = (int)(i - nc * (Ho * Wo));
  const int  ho = r / Wo;
  const int  wo = r - ho * Wo;
  const float* p = in + ((size_t)nc * H + 2 * ho) * W + 2 * wo;
  out[i] = fmaxf(fmaxf(p[0], p[1]), fmaxf(p[W], p[W + 1]));
}

// ---- per-(b,p,c) spatial mean: rows = B*P*C, each of length HW ----
__global__ void feat_mean(const float* __restrict__ in, float* __restrict__ out,
                          long nRows, int HW)
{
  long i = (long)blockIdx.x * blockDim.x + threadIdx.x;
  if (i >= nRows) return;
  const float* p = in + i * (long)HW;
  float s = 0.f;
  for (int j = 0; j < HW; ++j) s += p[j];
  out[i] = s * (1.0f / (float)HW);
}

// ---- scorer MLP (C -> 32 -> 1) + top-k indices per batch; 1 wave/block ----
__global__ void score_topk(const float* __restrict__ feat, // [B,P,C]
                           const float* __restrict__ sw1,  // [32,C]
                           const float* __restrict__ sb1,  // [32]
                           const float* __restrict__ sw2,  // [1,32]
                           const float* __restrict__ sb2,  // [1]
                           int* __restrict__ idxOut,       // [B,keep]
                           int P, int C, int keep)
{
  __shared__ float s_scores[16];
  const int b = blockIdx.x;
  const int t = threadIdx.x; // 0..31

  for (int p = 0; p < P; ++p) {
    const float* f    = feat + ((size_t)b * P + p) * C;
    const float* wrow = sw1 + (size_t)t * C;
    float acc = sb1[t];
    for (int ci = 0; ci < C; ++ci) acc += wrow[ci] * f[ci];
    acc = fmaxf(acc, 0.0f);
    float v = acc * sw2[t];
    for (int off = 16; off > 0; off >>= 1)
      v += __shfl_down(v, off, 32);
    if (t == 0) s_scores[p] = v + sb2[0];
  }
  __syncthreads();
  if (t == 0) {
    for (int j = 0; j < keep; ++j) {
      float best = -3.4e38f; int bi = 0;
      for (int p = 0; p < P; ++p)
        if (s_scores[p] > best) { best = s_scores[p]; bi = p; } // lowest idx on tie
      idxOut[b * keep + j] = bi;
      s_scores[bi] = -3.4e38f;
    }
  }
}

// ---- plain gather of kept paths (for the FC flat input, NCHW) ----
__global__ void gather_paths(const float* __restrict__ in, const int* __restrict__ idx,
                             float* __restrict__ out, int B, int P, int keep, long chw)
{
  long i = (long)blockIdx.x * blockDim.x + threadIdx.x;
  const long tot = (long)B * keep * chw;
  if (i >= tot) return;
  const long b   = i / ((long)keep * chw);
  const long r   = i - b * ((long)keep * chw);
  const int  j   = (int)(r / chw);
  const long off = r - (long)j * chw;
  out[i] = in[((long)b * P + idx[b * keep + j]) * chw + off];
}

// ---- final: mean over kept paths (ld=16 padded), * output_scale ----
__global__ void final_mean(const float* __restrict__ fin, float* __restrict__ out,
                           const float* __restrict__ scale, int B, int keep, int O)
{
  int i = blockIdx.x * blockDim.x + threadIdx.x;
  if (i >= B * O) return;
  const int b = i / O, o = i - b * O;
  float s = 0.f;
  for (int p = 0; p < keep; ++p) s += fin[(size_t)(b * keep + p) * 16 + o];
  out[i] = s * (1.0f / (float)keep) * scale[0];
}

// =========================================================================
extern "C" void kernel_launch(void* const* d_in, const int* in_sizes, int n_in,
                              void* d_out, int out_size, void* d_ws, size_t ws_size,
                              hipStream_t stream)
{
  (void)in_sizes; (void)n_in; (void)out_size; (void)ws_size;

  const float* x    = (const float*)d_in[0];
  const float* w1   = (const float*)d_in[1];
  const float* b1   = (const float*)d_in[2];
  const float* w2   = (const float*)d_in[3];
  const float* b2   = (const float*)d_in[4];
  const float* w3   = (const float*)d_in[5];
  const float* b3   = (const float*)d_in[6];
  const float* s1w1 = (const float*)d_in[7];
  const float* s1b1 = (const float*)d_in[8];
  const float* s1w2 = (const float*)d_in[9];
  const float* s1b2 = (const float*)d_in[10];
  const float* s2w1 = (const float*)d_in[11];
  const float* s2b1 = (const float*)d_in[12];
  const float* s2w2 = (const float*)d_in[13];
  const float* s2b2 = (const float*)d_in[14];
  const float* s3w1 = (const float*)d_in[15];
  const float* s3b1 = (const float*)d_in[16];
  const float* s3w2 = (const float*)d_in[17];
  const float* s3b2 = (const float*)d_in[18];
  const float* fc1w = (const float*)d_in[19];
  const float* fc1b = (const float*)d_in[20];
  const float* fc2w = (const float*)d_in[21];
  const float* fc2b = (const float*)d_in[22];
  const float* osc  = (const float*)d_in[23];
  float* out = (float*)d_out;

  // ---- workspace carve ----
  char* ws = (char*)d_ws;
  size_t off = 0;
  auto carve = [&](size_t bytes) -> void* {
    void* p = ws + off;
    off += (bytes + 255) & ~(size_t)255;
    return p;
  };
  float* bufA  = (float*)carve(402653184);  // conv outputs (max 100.7M f32)
  float* bufB  = (float*)carve(100663296);  // pooled outputs (max 25.2M f32)
  char*  bufCD = (char*)carve(67108864);    // padded f16 inputs OR fc flat (time-shared)
  _Float16* w1f   = (_Float16*)carve((size_t)192 * 288 * 2);   // CinP=32
  _Float16* w2f   = (_Float16*)carve((size_t)384 * 576 * 2);   // CinP=64
  _Float16* w3f   = (_Float16*)carve((size_t)768 * 1152 * 2);  // CinP=128
  _Float16* fc1wf = (_Float16*)carve((size_t)512 * 4096 * 2);
  _Float16* fc2wf = (_Float16*)carve((size_t)16 * 512 * 2);
  float*    fc2bp = (float*)carve(16 * 4);
  int*      idxb  = (int*)carve((size_t)512 * 8 * 4);

  _Float16* padIn = (_Float16*)bufCD;       // padded NHWC conv inputs (<= 52.5 MB)
  float*    flatF = (float*)bufCD;          // stage-3 flat gather [4096,4096] f32

  // sub-buffers aliasing bufA only while its previous contents are dead
  float*    feat   = (float*)((char*)bufA + 0);              // <= 6.3 MB
  _Float16* flat16 = (_Float16*)((char*)bufA + (16u << 20)); // 33.6 MB
  _Float16* fc1o   = (_Float16*)((char*)bufA + (56u << 20)); // 4.2 MB
  float*    fc2o   = (float*)((char*)bufA + (64u << 20));    // 0.26 MB

  const int B = 512;
  auto nb = [](long n) { return (unsigned)((n + 255) / 256); };

  // ---- weight repack / conversion ----
  cvt_conv_w<<<nb((long)192 * 288),  256, 0, stream>>>(w1, w1f, 192, 3,   32);
  cvt_conv_w<<<nb((long)384 * 576),  256, 0, stream>>>(w2, w2f, 384, 64,  64);
  cvt_conv_w<<<nb((long)768 * 1152), 256, 0, stream>>>(w3, w3f, 768, 128, 128);
  cvt_pad_f16<<<nb((long)512 * 4096), 256, 0, stream>>>(fc1w, fc1wf, 512, 4096, 512, 4096);
  cvt_pad_f16<<<nb((long)16 * 512),   256, 0, stream>>>(fc2w, fc2wf, 10,  512,  16,  512);
  pad_bias_f32<<<1, 32, 0, stream>>>(fc2b, fc2bp, 10, 16);

  // ================= Stage 1: 512 imgs, 3(->32 pad)->192, 32x32 =================
  gather_pad_f16<<<nb((long)512 * 34 * 34 * 32), 256, 0, stream>>>(
      x, nullptr, padIn, B, 1, 1, 3, 32, 32, 32);
  {
    dim3 g((512u * 32 * 32) / 128, 192 / 16);
    conv3x3_wmma_t<32, 32, 32><<<g, 256, 16 * 288 * 2, stream>>>(w1f, b1, padIn, bufA, 192);
  }
  pool2x2<<<nb((long)512 * 192 * 16 * 16), 256, 0, stream>>>(bufA, bufB, (long)512 * 192, 32, 32);
  feat_mean<<<nb((long)B * 3 * 64), 256, 0, stream>>>(bufB, feat, (long)B * 3 * 64, 256);
  score_topk<<<B, 32, 0, stream>>>(feat, s1w1, s1b1, s1w2, s1b2, idxb, 3, 64, 2);

  // ================= Stage 2: 1024 imgs, 64->384, 16x16 =================
  gather_pad_f16<<<nb((long)512 * 2 * 18 * 18 * 64), 256, 0, stream>>>(
      bufB, idxb, padIn, B, 3, 2, 64, 64, 16, 16);
  {
    dim3 g((1024u * 16 * 16) / 128, 384 / 16);
    conv3x3_wmma_t<64, 16, 16><<<g, 256, 16 * 576 * 2, stream>>>(w2f, b2, padIn, bufA, 384);
  }
  pool2x2<<<nb((long)1024 * 384 * 8 * 8), 256, 0, stream>>>(bufA, bufB, (long)1024 * 384, 16, 16);
  feat_mean<<<nb((long)B * 6 * 128), 256, 0, stream>>>(bufB, feat, (long)B * 6 * 128, 64);
  score_topk<<<B, 32, 0, stream>>>(feat, s2w1, s2b1, s2w2, s2b2, idxb, 6, 128, 4);

  // ================= Stage 3: 2048 imgs, 128->768, 8x8 =================
  gather_pad_f16<<<nb((long)512 * 4 * 10 * 10 * 128), 256, 0, stream>>>(
      bufB, idxb, padIn, B, 6, 4, 128, 128, 8, 8);
  {
    dim3 g((2048u * 8 * 8) / 128, 768 / 16);
    conv3x3_wmma_t<128, 8, 8><<<g, 256, 16 * 1152 * 2, stream>>>(w3f, b3, padIn, bufA, 768);
  }
  pool2x2<<<nb((long)2048 * 768 * 4 * 4), 256, 0, stream>>>(bufA, bufB, (long)2048 * 768, 8, 8);
  feat_mean<<<nb((long)B * 12 * 256), 256, 0, stream>>>(bufB, feat, (long)B * 12 * 256, 16);
  score_topk<<<B, 32, 0, stream>>>(feat, s3w1, s3b1, s3w2, s3b2, idxb, 12, 256, 8);

  // ================= Head =================
  // flat gather [512,8,4096] f32 (bufCD free: conv3 already consumed padIn)
  gather_paths<<<nb((long)B * 8 * 256 * 16), 256, 0, stream>>>(
      bufB, idxb, flatF, B, 12, 8, (long)256 * 16);
  cvt_pad_f16<<<nb((long)4096 * 4096), 256, 0, stream>>>(flatF, flat16, 4096, 4096, 4096, 4096);
  { // fc1: [4096,4096] x [512,4096]^T -> relu -> f16 [4096,512]
    dim3 g(512 / 16, 4096 / (16 * 8));
    gemm_nt_wmma<<<g, 256, 0, stream>>>(flat16, fc1wf, fc1b, nullptr, fc1o, 4096, 512, 4096, 1);
  }
  { // fc2: [4096,512] x [16,512]^T -> f32 [4096,16] (N padded 10->16)
    dim3 g(16 / 16, 4096 / (16 * 8));
    gemm_nt_wmma<<<g, 256, 0, stream>>>(fc1o, fc2wf, fc2bp, fc2o, nullptr, 4096, 16, 512, 0);
  }
  final_mean<<<nb((long)B * 10), 256, 0, stream>>>(fc2o, out, osc, B, 8, 10);
}
